// dividedSpaceTimeAttention_19877108646557
// MI455X (gfx1250) — matrix operand order repaired and checked by reading, not compiled, not run on hardware
//
#include <hip/hip_runtime.h>
#include <hip/hip_bf16.h>
#include <math.h>

// ---------------- problem constants ----------------
#define Bn   8
#define NHn  12
#define DIMn 768
#define NPn  196
#define NFn  16
#define D3n  192
#define SEQn 3136
#define TOKn 3137
#define C3n  2304                      // 3*DIM = NH*D3
#define SCALEf (1.0f/96.0f)            // 1/(sqrt(64)*12)

typedef unsigned short u16;
typedef unsigned int   u32;
typedef unsigned long long u64;
typedef __attribute__((ext_vector_type(16))) __bf16 v16bf;
typedef __attribute__((ext_vector_type(8)))  float  v8f;

// ---------------- CDNA5 async global->LDS support (guarded) ----------------
#ifndef HAVE_ASYNC_LDS
# if defined(__has_builtin)
#  if __has_builtin(__builtin_amdgcn_global_load_async_to_lds_b128)
#   define HAVE_ASYNC_LDS 1
#  endif
# endif
#endif
#ifndef HAVE_ASYNC_LDS
# define HAVE_ASYNC_LDS 0
#endif

// exact parameter pointee type per clang diagnostic: int __vector_size__(16)
typedef int v4i __attribute__((__vector_size__(16)));
typedef __attribute__((address_space(1))) v4i* gv4ip;   // global int4*
typedef __attribute__((address_space(3))) v4i* lv4ip;   // LDS int4* (32-bit)

__device__ __forceinline__ void async_copy_b128(const u16* g, u16* l) {
#if HAVE_ASYNC_LDS
  // generic->LDS addrspacecast on amdgcn == low-32-bit truncation (flat-LDS aperture)
  __builtin_amdgcn_global_load_async_to_lds_b128(
      (gv4ip)(u64)(g), (lv4ip)(u32)(u64)(l), 0, 0);
#else
  (void)g; (void)l;
#endif
}
__device__ __forceinline__ void wait_async0() {
#if HAVE_ASYNC_LDS
# if defined(__has_builtin) && __has_builtin(__builtin_amdgcn_s_wait_asynccnt)
  __builtin_amdgcn_s_wait_asynccnt(0);
# else
  asm volatile("s_wait_asynccnt 0x0" ::: "memory");
# endif
#endif
}

// ---------------- bf16 helpers ----------------
__device__ __forceinline__ float bflo(u32 u) { return __uint_as_float(u << 16); }
__device__ __forceinline__ float bfhi(u32 u) { return __uint_as_float(u & 0xFFFF0000u); }
__device__ __forceinline__ float b2f(u16 v)  { return __uint_as_float(((u32)v) << 16); }
__device__ __forceinline__ u16   f2b(float f) {
  u32 x = __float_as_uint(f);
  return (u16)((x + 0x7FFFu + ((x >> 16) & 1u)) >> 16);   // RNE
}
__device__ __forceinline__ v8f wmma_bf16(v16bf a, v16bf b, v8f c) {
  return __builtin_amdgcn_wmma_f32_16x16x32_bf16(false, a, false, b, (short)0, c, false, false);
}

// ---------------- fp32 -> bf16 conversion ----------------
__global__ void k_f32_to_bf16(const float* __restrict__ in, u16* __restrict__ out, size_t n) {
  size_t i = (size_t)blockIdx.x * blockDim.x + threadIdx.x;
  if (i < n) out[i] = f2b(in[i]);
}

// ---------------- generic bf16 WMMA GEMM: C(bf16)[M,N] = A[M,K] @ B[K,N] + bias ----
// 256 threads = 8 waves; block tile 128x64; wave tile 32x32 (4 WMMA / k-step);
// double-buffered LDS; A tile staged via GLOBAL_LOAD_ASYNC_TO_LDS_B128 when available.
#define BM 128
#define BN 64
#define BK 32

__global__ __launch_bounds__(256) void k_gemm_bf16(
    const u16* __restrict__ A, const u16* __restrict__ Bm,
    const float* __restrict__ bias, u16* __restrict__ Cb,
    int M, int N, int K)
{
  __shared__ u16 lA[2][BM * BK];   // [m][k]
  __shared__ u16 lB[2][BN * BK];   // [n][k]  (B transposed so k is contiguous)
  const int tid  = threadIdx.x;
  const int wave = tid >> 5;
  const int lane = tid & 31;
  const int l    = lane & 15;
  const int h    = lane >> 4;
  const int m0   = blockIdx.y * BM;
  const int n0   = blockIdx.x * BN;
  const int wm   = (wave & 3) * 32;
  const int wn   = (wave >> 2) * 32;

  v8f acc00 = {}, acc01 = {}, acc10 = {}, acc11 = {};

  const int ar = tid >> 1;          // 0..127 (A row)
  const int ac = (tid & 1) * 16;    // 0,16   (A col base; 16 elems per thread)
  const int br = tid >> 3;          // 0..31  (B k-row)
  const int bc = (tid & 7) * 8;     // 0..56  (B n-col)
  const bool fullM = (m0 + BM <= M);

  auto loadA = [&](int k0, int buf) {
    const int gm = m0 + ar;
    if (fullM && HAVE_ASYNC_LDS) {
      const u16* g = A + (size_t)gm * K + k0 + ac;
      u16* d = &lA[buf][ar * BK + ac];
      async_copy_b128(g, d);
      async_copy_b128(g + 8, d + 8);
    } else {
      int4 a0 = make_int4(0, 0, 0, 0), a1 = a0;
      if (gm < M) {
        const int4* g = (const int4*)(A + (size_t)gm * K + k0 + ac);
        a0 = g[0]; a1 = g[1];
      }
      *(int4*)&lA[buf][ar * BK + ac]     = a0;
      *(int4*)&lA[buf][ar * BK + ac + 8] = a1;
    }
  };
  auto loadB = [&](int k0, int buf) {
    int4 bvv = *(const int4*)(Bm + (size_t)(k0 + br) * N + n0 + bc);
    const u16* p = (const u16*)&bvv;
    #pragma unroll
    for (int i = 0; i < 8; ++i) lB[buf][(bc + i) * BK + br] = p[i];
    if (k0 + 2 * BK < K)         // keep a global_prefetch_b8 two tiles ahead
      __builtin_prefetch(Bm + (size_t)(k0 + 2 * BK + br) * N + n0 + bc, 0, 1);
  };
  auto compute = [&](int buf) {
    union F { v16bf v; u32 u[8]; } fa0, fa1, fb0, fb1;
    #pragma unroll
    for (int g = 0; g < 8; ++g) {
      const int ka = 16 * (g >> 2) + 8 * h + 2 * (g & 3);   // A 16x32 bf16 layout
      const int kb = 16 * (g >> 2) + 8 * h + 2 * (g & 3);   // B 32x16 bf16 layout (same k order as A)
      fa0.u[g] = *(const u32*)&lA[buf][(wm + l) * BK + ka];
      fa1.u[g] = *(const u32*)&lA[buf][(wm + 16 + l) * BK + ka];
      fb0.u[g] = *(const u32*)&lB[buf][(wn + l) * BK + kb];
      fb1.u[g] = *(const u32*)&lB[buf][(wn + 16 + l) * BK + kb];
    }
    acc00 = wmma_bf16(fa0.v, fb0.v, acc00);
    acc01 = wmma_bf16(fa0.v, fb1.v, acc01);
    acc10 = wmma_bf16(fa1.v, fb0.v, acc10);
    acc11 = wmma_bf16(fa1.v, fb1.v, acc11);
  };

  loadA(0, 0); loadB(0, 0);
  wait_async0();
  __syncthreads();

  int buf = 0;
  for (int k0 = BK; k0 < K; k0 += BK) {
    loadA(k0, buf ^ 1);          // stage next tile (ASYNCcnt path for A)
    loadB(k0, buf ^ 1);
    compute(buf);                // 4 WMMAs on current tile
    wait_async0();
    __syncthreads();
    buf ^= 1;
  }
  compute(buf);

  const float b0 = bias[n0 + wn + l];
  const float b1 = bias[n0 + wn + 16 + l];
  #pragma unroll
  for (int j = 0; j < 8; ++j) {                 // C layout: m = 8h + j, n = lane&15
    const int ma = m0 + wm + 8 * h + j;         // rows of acc0x
    const int mb = ma + 16;                     // rows of acc1x
    if (ma < M) {
      Cb[(size_t)ma * N + n0 + wn + l]      = f2b(acc00[j] + b0);
      Cb[(size_t)ma * N + n0 + wn + 16 + l] = f2b(acc01[j] + b1);
    }
    if (mb < M) {
      Cb[(size_t)mb * N + n0 + wn + l]      = f2b(acc10[j] + b0);
      Cb[(size_t)mb * N + n0 + wn + 16 + l] = f2b(acc11[j] + b1);
    }
  }
}

// ---------------- cls attention: 1 query vs 3137 keys per (b,h) (pre-RoPE) ------
__global__ __launch_bounds__(256) void k_cls_attn(
    const u16* __restrict__ Qp, const u16* __restrict__ Kp, const u16* __restrict__ Vp,
    float* __restrict__ clsrow)
{
  __shared__ float qv[D3n];
  __shared__ float sc[TOKn];
  __shared__ float red[256];
  const int b = blockIdx.y, hh = blockIdx.x, tid = threadIdx.x;
  const size_t base = (size_t)b * TOKn * C3n + (size_t)hh * D3n;

  if (tid < D3n) qv[tid] = b2f(Qp[base + tid]);   // query = token 0 (un-roped)
  __syncthreads();

  float lmax = -1e30f;
  for (int t = tid; t < TOKn; t += 256) {
    const u32* kp = (const u32*)(Kp + base + (size_t)t * C3n);
    float s = 0.f;
    #pragma unroll 4
    for (int i = 0; i < D3n / 2; ++i) {
      const u32 u = kp[i];
      s = fmaf(qv[2 * i], bflo(u), s);
      s = fmaf(qv[2 * i + 1], bfhi(u), s);
    }
    s *= SCALEf;
    sc[t] = s;
    lmax = fmaxf(lmax, s);
  }
  red[tid] = lmax; __syncthreads();
  for (int r = 128; r > 0; r >>= 1) { if (tid < r) red[tid] = fmaxf(red[tid], red[tid + r]); __syncthreads(); }
  const float m = red[0]; __syncthreads();

  float lsum = 0.f;
  for (int t = tid; t < TOKn; t += 256) { float e = __expf(sc[t] - m); sc[t] = e; lsum += e; }
  red[tid] = lsum; __syncthreads();
  for (int r = 128; r > 0; r >>= 1) { if (tid < r) red[tid] += red[tid + r]; __syncthreads(); }
  const float inv = 1.f / red[0]; __syncthreads();

  if (tid < D3n) {
    float acc = 0.f;
    for (int t = 0; t < TOKn; ++t)
      acc = fmaf(sc[t], b2f(Vp[base + (size_t)t * C3n + tid]), acc);
    clsrow[(size_t)b * C3n + hh * D3n + tid] = acc * inv;
  }
}

// ---------------- RoPE (in-place, body rows only, first 64 dims of each head) ----
__global__ void k_rope(u16* __restrict__ Qp, u16* __restrict__ Vp) {
  const size_t total = (size_t)Bn * SEQn * NHn * 32;
  size_t i = (size_t)blockIdx.x * blockDim.x + threadIdx.x;
  if (i >= total) return;
  const int c = (int)(i & 31); size_t t = i >> 5;
  const int hh = (int)(t % NHn); t /= NHn;
  const int s  = (int)(t % SEQn);
  const int b  = (int)(t / SEQn);
  const float invf = __powf(10000.0f, -((float)(2 * c)) / 64.0f);
  const float ang = (float)s * invf;
  const float cs = __cosf(ang), sn = __sinf(ang);
  const size_t off = ((size_t)(b * TOKn + 1 + s)) * C3n + hh * D3n + 2 * c;
  { const float x1 = b2f(Qp[off]), x2 = b2f(Qp[off + 1]);
    Qp[off] = f2b(x1 * cs - x2 * sn); Qp[off + 1] = f2b(x2 * cs + x1 * sn); }
  { const float x1 = b2f(Vp[off]), x2 = b2f(Vp[off + 1]);
    Vp[off] = f2b(x1 * cs - x2 * sn); Vp[off + 1] = f2b(x2 * cs + x1 * sn); }
}

// ---------------- temporal attention: 16x16 per (b,h,patch), summed over f -------
__global__ __launch_bounds__(256) void k_temporal(
    const u16* __restrict__ Qp, const u16* __restrict__ Vp, const u16* __restrict__ Kp,
    u16* __restrict__ tmpc)
{
  __shared__ float S[16][16];
  __shared__ float colw[16];
  const int p  = blockIdx.x + 1;      // patches 1..195
  const int hh = blockIdx.y;
  const int b  = blockIdx.z;
  const int tid = threadIdx.x;
  const size_t hoff = (size_t)hh * D3n;
  auto row = [&](int f) -> size_t { return (size_t)(b * TOKn + 1 + f * NPn + p); };

  { const int f = tid >> 4, g = tid & 15;
    const u32* qa = (const u32*)(Qp + row(f) * C3n + hoff);
    const u32* kb = (const u32*)(Vp + row(g) * C3n + hoff);   // k = roped x@Wv
    float s = 0.f;
    for (int i = 0; i < D3n / 2; ++i) {
      const u32 ua = qa[i], ub = kb[i];
      s = fmaf(bflo(ua), bflo(ub), s);
      s = fmaf(bfhi(ua), bfhi(ub), s);
    }
    S[f][g] = s * SCALEf;
  }
  __syncthreads();
  if (tid < 16) {
    float m = -1e30f;
    for (int g = 0; g < 16; ++g) m = fmaxf(m, S[tid][g]);
    float l = 0.f;
    for (int g = 0; g < 16; ++g) { const float e = __expf(S[tid][g] - m); S[tid][g] = e; l += e; }
    const float inv = 1.f / l;
    for (int g = 0; g < 16; ++g) S[tid][g] *= inv;
  }
  __syncthreads();
  if (tid < 16) {
    float s = 0.f;
    for (int f = 0; f < 16; ++f) s += S[f][tid];
    colw[tid] = s;                                  // sum over f of P
  }
  __syncthreads();
  if (tid < D3n) {
    float acc = 0.f;
    for (int g = 0; g < 16; ++g)
      acc = fmaf(colw[g], b2f(Kp[row(g) * C3n + hoff + tid]), acc);  // v = x@Wk
    tmpc[((size_t)(b * 195) + (p - 1)) * C3n + hoff + tid] = f2b(acc);
  }
}

// ---------------- final attention + query-sum (w-vector trick) -------------------
__global__ __launch_bounds__(256) void k_final_attn(
    const u16* __restrict__ Q2, const u16* __restrict__ K2, const u16* __restrict__ V2,
    float* __restrict__ t2u)
{
  __shared__ float w[NPn];
  __shared__ float red[256];
  const int xi = blockIdx.x;
  const int hh = blockIdx.y;
  const int b  = blockIdx.z;
  const int tid = threadIdx.x;
  const int Nk = (xi == 0) ? 196 : 16;
  const size_t hoff = (size_t)hh * D3n;

  if (tid < Nk) w[tid] = 0.f;
  __syncthreads();

  int krow = 0;
  if (tid < Nk) krow = (xi == 0) ? (tid % 195) : ((xi + tid) % 195);
  const u32* kp = (const u32*)(K2 + ((size_t)(b * 195 + krow)) * C3n + hoff);

  for (int q = 0; q < NPn; ++q) {
    const int qrow = (xi + q) % 195;
    const u32* qp = (const u32*)(Q2 + ((size_t)(b * 195 + qrow)) * C3n + hoff);
    float s = -1e30f;
    if (tid < Nk) {
      float acc = 0.f;
      for (int i = 0; i < D3n / 2; ++i) {
        const u32 ua = qp[i], ub = kp[i];
        acc = fmaf(bflo(ua), bflo(ub), acc);
        acc = fmaf(bfhi(ua), bfhi(ub), acc);
      }
      s = acc * SCALEf;
    }
    red[tid] = s; __syncthreads();
    for (int r = 128; r > 0; r >>= 1) { if (tid < r) red[tid] = fmaxf(red[tid], red[tid + r]); __syncthreads(); }
    const float m = red[0]; __syncthreads();
    const float e = (tid < Nk) ? __expf(s - m) : 0.f;
    red[tid] = e; __syncthreads();
    for (int r = 128; r > 0; r >>= 1) { if (tid < r) red[tid] += red[tid + r]; __syncthreads(); }
    const float inv = 1.f / red[0]; __syncthreads();
    if (tid < Nk) w[tid] += e * inv;
    __syncthreads();
  }

  if (tid < D3n) {
    float acc = 0.f;
    for (int t = 0; t < Nk; ++t) {
      const int kr = (xi == 0) ? (t % 195) : ((xi + t) % 195);
      acc = fmaf(w[t], b2f(V2[((size_t)(b * 195 + kr)) * C3n + hoff + tid]), acc);
    }
    t2u[((size_t)(b * 16 + xi)) * C3n + hoff + tid] = acc;
  }
}

// ---------------- final projection of the 17 unique rows per batch ---------------
__global__ __launch_bounds__(256) void k_final_proj(
    const float* __restrict__ clsrow, const float* __restrict__ t2u,
    const float* __restrict__ Wf, const float* __restrict__ bf,
    float* __restrict__ U)
{
  __shared__ float sa[C3n];
  const int r = blockIdx.x;           // 0 = cls, 1..16 = t2u rows
  const int b = blockIdx.y;
  const int tid = threadIdx.x;
  const float* arow = (r == 0) ? (clsrow + (size_t)b * C3n)
                               : (t2u + ((size_t)(b * 16 + r - 1)) * C3n);
  for (int k = tid; k < C3n; k += 256) sa[k] = arow[k];
  __syncthreads();
  for (int n = tid; n < DIMn; n += 256) {
    float acc = bf[n];
    for (int k = 0; k < C3n; ++k) acc = fmaf(sa[k], Wf[(size_t)k * DIMn + n], acc);
    U[((size_t)(b * 17 + r)) * DIMn + n] = acc;
  }
}

// ---------------- broadcast the 17 unique rows into the full output --------------
__global__ void k_broadcast(const float* __restrict__ U, float* __restrict__ out, size_t total) {
  size_t i = (size_t)blockIdx.x * blockDim.x + threadIdx.x;
  if (i >= total) return;
  const int c = (int)(i % DIMn);
  const size_t t = i / DIMn;
  const int s = (int)(t % TOKn);
  const int b = (int)(t / TOKn);
  const int r = (s == 0) ? 0 : 1 + ((s - 1) & 15);
  out[i] = U[((size_t)(b * 17 + r)) * DIMn + c];
}

// ---------------- host orchestration ----------------
extern "C" void kernel_launch(void* const* d_in, const int* in_sizes, int n_in,
                              void* d_out, int out_size, void* d_ws, size_t ws_size,
                              hipStream_t stream) {
  (void)in_sizes; (void)n_in; (void)out_size; (void)ws_size;
  const float* x  = (const float*)d_in[0];
  const float* Wq = (const float*)d_in[1];
  const float* bq = (const float*)d_in[2];
  const float* Wk = (const float*)d_in[3];
  const float* bk = (const float*)d_in[4];
  const float* Wv = (const float*)d_in[5];
  const float* bv = (const float*)d_in[6];
  const float* Wt = (const float*)d_in[7];
  const float* bt = (const float*)d_in[8];
  const float* Wf = (const float*)d_in[9];
  const float* bf = (const float*)d_in[10];
  float* out = (float*)d_out;

  char* ws = (char*)d_ws;
  size_t off = 0;
  auto alloc = [&](size_t bytes) -> char* {
    char* p = ws + off;
    off = (off + bytes + 255) & ~(size_t)255;
    return p;
  };
  const size_t MX   = (size_t)Bn * TOKn;   // 25096 token rows
  const size_t M195 = (size_t)Bn * 195;    // 1560 unique temporal rows

  u16* xb   = (u16*)alloc(MX * DIMn * 2);
  u16* Wqb  = (u16*)alloc((size_t)DIMn * C3n * 2);
  u16* Wkb  = (u16*)alloc((size_t)DIMn * C3n * 2);
  u16* Wvb  = (u16*)alloc((size_t)DIMn * C3n * 2);
  u16* Wtb  = (u16*)alloc((size_t)C3n * DIMn * 2);
  u16* Qp   = (u16*)alloc(MX * C3n * 2);
  u16* Kp   = (u16*)alloc(MX * C3n * 2);
  u16* Vp   = (u16*)alloc(MX * C3n * 2);
  u16* tmpc = (u16*)alloc(M195 * C3n * 2);
  u16* TI   = (u16*)alloc(M195 * DIMn * 2);
  u16* Q2   = (u16*)alloc(M195 * C3n * 2);
  u16* K2v  = (u16*)alloc(M195 * C3n * 2);
  u16* V2k  = (u16*)alloc(M195 * C3n * 2);
  float* clsrow = (float*)alloc((size_t)Bn * C3n * 4);
  float* t2u    = (float*)alloc((size_t)Bn * 16 * C3n * 4);
  float* U      = (float*)alloc((size_t)Bn * 17 * DIMn * 4);

  // 1. fp32 -> bf16 conversions
  { const size_t n = MX * DIMn;
    k_f32_to_bf16<<<(unsigned)((n + 255) / 256), 256, 0, stream>>>(x, xb, n); }
  { const size_t n = (size_t)DIMn * C3n;
    const unsigned g = (unsigned)((n + 255) / 256);
    k_f32_to_bf16<<<g, 256, 0, stream>>>(Wq, Wqb, n);
    k_f32_to_bf16<<<g, 256, 0, stream>>>(Wk, Wkb, n);
    k_f32_to_bf16<<<g, 256, 0, stream>>>(Wv, Wvb, n);
    k_f32_to_bf16<<<g, 256, 0, stream>>>(Wt, Wtb, n); }

  // 2. input projections (WMMA): dominant 266 GFLOP
  { dim3 g(C3n / BN, (unsigned)((MX + BM - 1) / BM));
    k_gemm_bf16<<<g, 256, 0, stream>>>(xb, Wqb, bq, Qp, (int)MX, C3n, DIMn);
    k_gemm_bf16<<<g, 256, 0, stream>>>(xb, Wkb, bk, Kp, (int)MX, C3n, DIMn);
    k_gemm_bf16<<<g, 256, 0, stream>>>(xb, Wvb, bv, Vp, (int)MX, C3n, DIMn); }

  // 3. cls attention (must run before in-place RoPE: v_cls is un-roped x@Wv)
  { dim3 g(NHn, Bn); k_cls_attn<<<g, 256, 0, stream>>>(Qp, Kp, Vp, clsrow); }

  // 4. RoPE on body rows of Qp (q) and Vp (k)
  { const size_t n = (size_t)Bn * SEQn * NHn * 32;
    k_rope<<<(unsigned)((n + 255) / 256), 256, 0, stream>>>(Qp, Vp); }

  // 5. temporal attention over frames, summed -> 195 unique rows/batch
  { dim3 g(195, NHn, Bn); k_temporal<<<g, 256, 0, stream>>>(Qp, Vp, Kp, tmpc); }

  // 6. TI = tmpc @ Wt + bt (WMMA)
  { dim3 g(DIMn / BN, (unsigned)((M195 + BM - 1) / BM));
    k_gemm_bf16<<<g, 256, 0, stream>>>(tmpc, Wtb, bt, TI, (int)M195, DIMn, C3n); }

  // 7. stage-2 projections (WMMA) — note reference's swap: k2<-Wv, v2<-Wk
  { dim3 g(C3n / BN, (unsigned)((M195 + BM - 1) / BM));
    k_gemm_bf16<<<g, 256, 0, stream>>>(TI, Wqb, bq, Q2,  (int)M195, C3n, DIMn);
    k_gemm_bf16<<<g, 256, 0, stream>>>(TI, Wvb, bv, K2v, (int)M195, C3n, DIMn);
    k_gemm_bf16<<<g, 256, 0, stream>>>(TI, Wkb, bk, V2k, (int)M195, C3n, DIMn); }

  // 8. final attention with query-sum reduction
  { dim3 g(16, NHn, Bn); k_final_attn<<<g, 256, 0, stream>>>(Q2, K2v, V2k, t2u); }

  // 9. project the 17 unique rows per batch through Wf
  { dim3 g(17, Bn); k_final_proj<<<g, 256, 0, stream>>>(clsrow, t2u, Wf, bf, U); }

  // 10. broadcast unique rows into the full (8,3137,768) output
  { const size_t total = (size_t)Bn * TOKn * DIMn;
    k_broadcast<<<(unsigned)((total + 255) / 256), 256, 0, stream>>>(U, out, total); }
}
